// CorssCasualAttention_48790828482896
// MI455X (gfx1250) — compile-verified, matrix operand order checked by reading, not run on hardware
//
#include <hip/hip_runtime.h>
#include <math.h>

// ---------------------------------------------------------------------------
// Cross block-causal attention for MI455X (gfx1250), wave32 + WMMA f16.
// Shapes: v=8 b=4 c=512 h=w=16 -> T=2048, 8 heads x d=64, causal block = 256.
// GEMMs stage B through LDS with gfx1250 async global->LDS loads (ASYNCcnt).
// ---------------------------------------------------------------------------

typedef __attribute__((ext_vector_type(16))) _Float16 v16h;
typedef __attribute__((ext_vector_type(8)))  _Float16 v8h;
typedef __attribute__((ext_vector_type(8)))  float    v8f;
typedef __attribute__((__vector_size__(16))) int      v4i_;
typedef __attribute__((address_space(1)))    v4i_     gv4i;   // global <4 x i32>
typedef __attribute__((address_space(3)))    v4i_     lv4i;   // LDS    <4 x i32>

#define VDIM   8
#define BDIM   4
#define CDIM   512
#define TDIM   2048
#define NHEAD  8
#define HDIM   64
#define MROWS  (BDIM * TDIM)   // 8192 GEMM rows

#if defined(__HIP_DEVICE_COMPILE__) && \
    __has_builtin(__builtin_amdgcn_global_load_async_to_lds_b128) && \
    __has_builtin(__builtin_amdgcn_s_wait_asynccnt)
#define CCA_ASYNC_LDS 1
#else
#define CCA_ASYNC_LDS 0
#endif

// ---- WMMA fragment loader ---------------------------------------------------
// ISA 16-bit A/B layout for 16x16x32 (CDNA5 ISA):
//   lanes 0-15 : K = {0..7} in halves 0-7,  K = {16..23} in halves 8-15
//   lanes 16-31: K = {8..15} in halves 0-7, K = {24..31} in halves 8-15
// rowp = start of this row's 32-wide K slice; ko = (lane>>4)*8.
__device__ __forceinline__ v16h load_frag(const _Float16* rowp, int ko) {
  v8h c0 = *(const v8h*)(rowp + ko);
  v8h c1 = *(const v8h*)(rowp + 16 + ko);
  v16h r;
#pragma unroll
  for (int i = 0; i < 8; ++i) { r[i] = c0[i]; r[i + 8] = c1[i]; }
  return r;
}

// ---- Pack [v,b,c,h,w] f32 -> [b,T,c] f16 -----------------------------------
__global__ void cca_pack_x(const float* __restrict__ src,
                           _Float16* __restrict__ dst, int total) {
  for (int i = blockIdx.x * blockDim.x + threadIdx.x; i < total;
       i += gridDim.x * blockDim.x) {
    int cc = i & (CDIM - 1);
    int t  = (i >> 9) & (TDIM - 1);
    int b  = i >> 20;                 // / (CDIM*TDIM)
    int vv = t >> 8;
    int hh = (t >> 4) & 15;
    int ww = t & 15;
    int s = (((vv * BDIM + b) * CDIM + cc) * 16 + hh) * 16 + ww;
    dst[i] = (_Float16)src[s];
  }
}

// ---- Pack W[K][N] f32 -> Wt[N][K] f16 --------------------------------------
__global__ void cca_pack_wT(const float* __restrict__ W,
                            _Float16* __restrict__ Wt, int K, int N) {
  int total = K * N;
  for (int i = blockIdx.x * blockDim.x + threadIdx.x; i < total;
       i += gridDim.x * blockDim.x) {
    int k = i / N;
    int n = i - k * N;
    Wt[(size_t)n * K + k] = (_Float16)W[i];
  }
}

// ---- WMMA GEMM: C[8192,N] = A[8192,512] * Bt^T + bias -----------------------
// Block = 4 waves = 128x64 output tile. Per K-step (32), the 64x32 B slab is
// async-copied to LDS once and shared by all waves; each wave owns a 32x64
// tile (8 f32 accumulators) so each B fragment feeds two WMMAs.
// mode 0: Q projection  -> qh[b*8+h][T][d] f16, scaled by 1/sqrt(d)
// mode 1: KV projection -> kh[b*8+h][T][d] f16 ; vT[b*8+h][d][T] f16
// mode 2: out projection-> f32 scatter to [v,b,c,h,w]
__global__ void cca_proj_gemm(const _Float16* __restrict__ A,
                              const _Float16* __restrict__ Bt,
                              const float* __restrict__ bias,
                              _Float16* __restrict__ dst16,
                              _Float16* __restrict__ vTo,
                              float* __restrict__ outf,
                              int Ncols, int mode) {
  __shared__ __align__(16) _Float16 blds[64 * 32];   // 4 KB B slab

  int tilesN = Ncols >> 6;
  int blkRow = blockIdx.x / tilesN;
  int n0 = (blockIdx.x - blkRow * tilesN) << 6;
  int tid = threadIdx.x;
  int wid = tid >> 5;
  int lane = tid & 31;
  int m0 = blkRow * 128 + wid * 32;      // this wave's 32-row strip

  int lr = lane & 15;          // row (A) / column-row (B) within tile
  int ko = (lane >> 4) << 3;   // K sub-chunk select per ISA layout

  v8f acc[2][4] = {};
  const _Float16* arow0 = A + (size_t)(m0 + lr) * CDIM;
  const _Float16* arow1 = A + (size_t)(m0 + 16 + lr) * CDIM;

  for (int k = 0; k < CDIM; k += 32) {
    __syncthreads();   // previous-iteration LDS reads complete
    // ---- stage B slab: rows n0..n0+63, K-slice k..k+31 (async -> LDS) ------
#pragma unroll
    for (int e = tid; e < 256; e += 128) {           // 256 x 16B = 4 KB
      const _Float16* gsrc = Bt + (size_t)(n0 + (e >> 2)) * CDIM + k + (e & 3) * 8;
      _Float16* ldst = &blds[e * 8];
#if CCA_ASYNC_LDS
      __builtin_amdgcn_global_load_async_to_lds_b128(
          (gv4i*)gsrc, (lv4i*)ldst, 0, 0);
#else
      *(v8h*)ldst = *(const v8h*)gsrc;
#endif
    }
#if CCA_ASYNC_LDS
    __builtin_amdgcn_s_wait_asynccnt(0);
#endif
    // ---- A fragments (direct, L2-resident) + prefetch of next K slice ------
    if (k + 32 < CDIM) {
      __builtin_prefetch(arow0 + k + 32, 0, 1);
      __builtin_prefetch(arow1 + k + 32, 0, 1);
    }
    v16h a0 = load_frag(arow0 + k, ko);
    v16h a1 = load_frag(arow1 + k, ko);
    __syncthreads();   // B slab visible to all waves
#pragma unroll
    for (int t = 0; t < 4; ++t) {
      v16h bf = load_frag(blds + (t * 16 + lr) * 32, ko);
      acc[0][t] = __builtin_amdgcn_wmma_f32_16x16x32_f16(
          false, a0, false, bf, (short)0, acc[0][t], false, false);
      acc[1][t] = __builtin_amdgcn_wmma_f32_16x16x32_f16(
          false, a1, false, bf, (short)0, acc[1][t], false, false);
    }
  }

  // C layout: VGPR r, lanes 0-15 -> row r, lanes 16-31 -> row r+8; col = lane&15
  int rbase = (lane >> 4) << 3;
#pragma unroll
  for (int t = 0; t < 4; ++t) {
    int n = n0 + t * 16 + lr;
    float bv = bias[n];
#pragma unroll
    for (int s2 = 0; s2 < 2; ++s2) {
#pragma unroll
      for (int r = 0; r < 8; ++r) {
        int m = m0 + s2 * 16 + rbase + r;
        float val = acc[s2][t][r] + bv;
        int b  = m >> 11;          // / TDIM
        int tt = m & (TDIM - 1);
        if (mode == 0) {
          int hh = n >> 6, dd = n & 63;
          dst16[((size_t)(b * NHEAD + hh) * TDIM + tt) * HDIM + dd] =
              (_Float16)(val * 0.125f);   // fold 1/sqrt(64)
        } else if (mode == 1) {
          if (n < CDIM) {
            int hh = n >> 6, dd = n & 63;
            dst16[((size_t)(b * NHEAD + hh) * TDIM + tt) * HDIM + dd] = (_Float16)val;
          } else {
            int nn = n - CDIM;
            int hh = nn >> 6, dd = nn & 63;
            vTo[((size_t)(b * NHEAD + hh) * HDIM + dd) * TDIM + tt] = (_Float16)val;
          }
        } else {
          int vv = tt >> 8, hh = (tt >> 4) & 15, ww = tt & 15;
          outf[(((size_t)(vv * BDIM + b) * CDIM + n) * 16 + hh) * 16 + ww] = val;
        }
      }
    }
  }
}

// ---- Flash attention: one wave per 16-row query tile ------------------------
__global__ void cca_attn(const _Float16* __restrict__ qh,
                         const _Float16* __restrict__ kh,
                         const _Float16* __restrict__ vT,
                         _Float16* __restrict__ ybuf) {
  __shared__ __align__(16) _Float16 plds[16 * 32];

  int bh = blockIdx.x >> 7;          // 0..31  (b*8 + head)
  int qt = blockIdx.x & 127;         // query tile
  int q0 = qt << 4;
  int kend = ((q0 >> 8) + 1) << 8;   // block-causal bound: (qblock+1)*256

  const _Float16* Q  = qh + (size_t)bh * TDIM * HDIM;
  const _Float16* Kh = kh + (size_t)bh * TDIM * HDIM;
  const _Float16* Vt = vT + (size_t)bh * HDIM * TDIM;

  int lane = threadIdx.x;
  int lr = lane & 15;
  int ko = (lane >> 4) << 3;

  // Q fragments for both d-slices (K = 0..31 and 32..63), reused all iterations
  v16h aq[2];
#pragma unroll
  for (int s = 0; s < 2; ++s)
    aq[s] = load_frag(Q + (size_t)(q0 + lr) * HDIM + s * 32, ko);

  v8f o[4] = {};
  float mi[8], li[8];
#pragma unroll
  for (int r = 0; r < 8; ++r) { mi[r] = -1e30f; li[r] = 0.0f; }

  for (int kk = 0; kk < kend; kk += 32) {
    // ---- S = Q K^T for 32 keys: two 16x16 f32 accumulators -----------------
    v8f s0 = {}, s1 = {};
#pragma unroll
    for (int s = 0; s < 2; ++s) {
      v16h bk0 = load_frag(Kh + (size_t)(kk + lr) * HDIM + s * 32, ko);
      v16h bk1 = load_frag(Kh + (size_t)(kk + 16 + lr) * HDIM + s * 32, ko);
      s0 = __builtin_amdgcn_wmma_f32_16x16x32_f16(false, aq[s], false, bk0,
                                                  (short)0, s0, false, false);
      s1 = __builtin_amdgcn_wmma_f32_16x16x32_f16(false, aq[s], false, bk1,
                                                  (short)0, s1, false, false);
    }

    // ---- online softmax (row stats live per VGPR slot, halves = row groups) -
#pragma unroll
    for (int r = 0; r < 8; ++r) {
      float v = fmaxf(s0[r], s1[r]);
      v = fmaxf(v, __shfl_xor(v, 1, 32));
      v = fmaxf(v, __shfl_xor(v, 2, 32));
      v = fmaxf(v, __shfl_xor(v, 4, 32));
      v = fmaxf(v, __shfl_xor(v, 8, 32));
      float mnew = fmaxf(mi[r], v);
      float alpha = __expf(mi[r] - mnew);
      mi[r] = mnew;
      float p0 = __expf(s0[r] - mnew);
      float p1 = __expf(s1[r] - mnew);
      float ps = p0 + p1;
      ps += __shfl_xor(ps, 1, 32);
      ps += __shfl_xor(ps, 2, 32);
      ps += __shfl_xor(ps, 4, 32);
      ps += __shfl_xor(ps, 8, 32);
      li[r] = li[r] * alpha + ps;
#pragma unroll
      for (int t = 0; t < 4; ++t) o[t][r] *= alpha;
      int prow = r + ((lane >> 4) << 3);
      plds[prow * 32 + lr] = (_Float16)p0;
      plds[prow * 32 + 16 + lr] = (_Float16)p1;
    }
    __syncthreads();   // single wave: compiler emits the s_wait_dscnt we need

    // ---- O += P(16x32) * V(32x64): V^T rows are contiguous B operands ------
    v16h ap = load_frag(plds + lr * 32, ko);
#pragma unroll
    for (int t = 0; t < 4; ++t) {
      v16h bv = load_frag(Vt + (size_t)(t * 16 + lr) * TDIM + kk, ko);
      o[t] = __builtin_amdgcn_wmma_f32_16x16x32_f16(false, ap, false, bv,
                                                    (short)0, o[t], false, false);
    }
    __syncthreads();
  }

  // ---- normalize and store to [b,T,c] f16 for the final projection ---------
  int b = bh >> 3;
  int hh = bh & 7;
  int rbase = (lane >> 4) << 3;
#pragma unroll
  for (int t = 0; t < 4; ++t) {
#pragma unroll
    for (int r = 0; r < 8; ++r) {
      int row = q0 + rbase + r;
      int cc = hh * HDIM + t * 16 + lr;
      ybuf[((size_t)b * TDIM + row) * CDIM + cc] = (_Float16)(o[t][r] / li[r]);
    }
  }
}

// ---------------------------------------------------------------------------
extern "C" void kernel_launch(void* const* d_in, const int* in_sizes, int n_in,
                              void* d_out, int out_size, void* d_ws, size_t ws_size,
                              hipStream_t stream) {
  const float* q   = (const float*)d_in[0];
  const float* kv  = (const float*)d_in[1];
  const float* Wq  = (const float*)d_in[2];
  const float* bq  = (const float*)d_in[3];
  const float* Wkv = (const float*)d_in[4];
  const float* bkv = (const float*)d_in[5];
  const float* Wp  = (const float*)d_in[6];
  const float* bp  = (const float*)d_in[7];
  float* out = (float*)d_out;

  char* ws = (char*)d_ws;
  _Float16* xq   = (_Float16*)(ws + 0);          //  8 MB  [b,T,c]
  _Float16* xkv  = (_Float16*)(ws + 8388608);    //  8 MB
  _Float16* WqT  = (_Float16*)(ws + 16777216);   // .5 MB  [N,K]
  _Float16* WkvT = (_Float16*)(ws + 17301504);   //  1 MB
  _Float16* WpT  = (_Float16*)(ws + 18350080);   // .5 MB
  _Float16* qh   = (_Float16*)(ws + 18874368);   //  8 MB  [bh,T,d] (pre-scaled)
  _Float16* kh   = (_Float16*)(ws + 27262976);   //  8 MB  [bh,T,d]
  _Float16* vT   = (_Float16*)(ws + 35651584);   //  8 MB  [bh,d,T]
  _Float16* ybuf = (_Float16*)(ws + 44040192);   //  8 MB  [b,T,c]

  const int totalX = BDIM * TDIM * CDIM;         // 4,194,304

  cca_pack_x<<<4096, 256, 0, stream>>>(q, xq, totalX);
  cca_pack_x<<<4096, 256, 0, stream>>>(kv, xkv, totalX);
  cca_pack_wT<<<1024, 256, 0, stream>>>(Wq, WqT, CDIM, CDIM);
  cca_pack_wT<<<2048, 256, 0, stream>>>(Wkv, WkvT, CDIM, 2 * CDIM);
  cca_pack_wT<<<1024, 256, 0, stream>>>(Wp, WpT, CDIM, CDIM);

  // Q: 8192 rows / 128 per block * (512/64) col tiles = 512 blocks
  cca_proj_gemm<<<512, 128, 0, stream>>>(xq, WqT, bq, qh, nullptr, nullptr,
                                         CDIM, 0);
  // KV: N=1024 -> 64 * 16 = 1024 blocks
  cca_proj_gemm<<<1024, 128, 0, stream>>>(xkv, WkvT, bkv, kh, vT, nullptr,
                                          2 * CDIM, 1);
  // attention: 32 (b,h) x 128 query tiles = 4096 waves, one per workgroup
  cca_attn<<<4096, 32, 0, stream>>>(qh, kh, vT, ybuf);
  // final projection + scatter to [v,b,c,h,w]
  cca_proj_gemm<<<512, 128, 0, stream>>>(ybuf, WpT, bp, nullptr, nullptr, out,
                                         CDIM, 2);
}
